// GCNPool_dynamic_19447611916445
// MI455X (gfx1250) — matrix operand order repaired and checked by reading, not compiled, not run on hardware
//
#include <hip/hip_runtime.h>
#include <hip/hip_bf16.h>

// ---------------------------------------------------------------------------
// GCNPool_dynamic forward for MI455X (gfx1250, wave32, WMMA)
// Heavy parts (graph diffusion GEMMs, mlp conv GEMM) -> v_wmma_f32_16x16x32_bf16
// ---------------------------------------------------------------------------

#define BB   8
#define CIN  32
#define CO   64
#define NN   2048
#define TT   12
#define L1D  8          // after dilated tconv
#define LL   6          // final temporal size
#define MD   4096       // B*CO*L1 = 8*64*8
#define C7   448        // 7*CO

typedef __attribute__((ext_vector_type(16))) __bf16          v16bf;
typedef __attribute__((ext_vector_type(8)))  __bf16          v8bf;
typedef __attribute__((ext_vector_type(8)))  float           v8f;
typedef __attribute__((ext_vector_type(8)))  unsigned short  v8u16;

__device__ __forceinline__ unsigned short f2bf(float f) {
    union { float f; unsigned u; } v; v.f = f;
    unsigned r = v.u + 0x7FFFu + ((v.u >> 16) & 1u);
    return (unsigned short)(r >> 16);
}

__device__ __forceinline__ v8f wmma_bf16(v16bf a, v16bf b, v8f c) {
    return __builtin_amdgcn_wmma_f32_16x16x32_bf16(false, a, false, b, (short)0, c,
                                                   false, false);
}

__device__ __forceinline__ float sigmoidf_(float x) {
    return 1.0f / (1.0f + __expf(-x));
}

// --------------------------- fp32 -> bf16 convert ---------------------------
__global__ void k_sup2bf(const float* __restrict__ s, unsigned short* __restrict__ d, int count) {
    int stride = gridDim.x * blockDim.x;
    for (int i = blockIdx.x * blockDim.x + threadIdx.x; i < count; i += stride)
        d[i] = f2bf(s[i]);
}

// --------------------------- gated dilated tconv ----------------------------
// x[B,32,N,12] -> H0_T[m][v] bf16, m=((b*64+c)*8+l), v=n;  h = tanh(x1)*sig(x2)
__global__ __launch_bounds__(256) void k_tconv(
    const float* __restrict__ x, const float* __restrict__ tw,
    const float* __restrict__ tb, unsigned short* __restrict__ H0)
{
    __shared__ float xs[CIN][3][64 + 1];
    int bid = blockIdx.x;
    int nb = (bid & 31) * 64; bid >>= 5;
    int l  = bid & 7;
    int b  = bid >> 3;

    for (int e = threadIdx.x; e < CIN * 3 * 64; e += 256) {
        int n = e & 63, rest = e >> 6;
        int k = rest % 3, ci = rest / 3;
        xs[ci][k][n] = x[((size_t)(b * CIN + ci) * NN + nb + n) * TT + l + 2 * k];
    }
    __syncthreads();

    int c  = threadIdx.x & 63;
    int nq = threadIdx.x >> 6;       // 0..3 -> 16 n each
    float a1[16], a2[16];
#pragma unroll
    for (int i = 0; i < 16; ++i) { a1[i] = tb[c]; a2[i] = tb[64 + c]; }

    for (int ci = 0; ci < CIN; ++ci)
#pragma unroll
        for (int k = 0; k < 3; ++k) {
            float w1 = tw[(c * CIN + ci) * 3 + k];
            float w2 = tw[((64 + c) * CIN + ci) * 3 + k];
#pragma unroll
            for (int i = 0; i < 16; ++i) {
                float xv = xs[ci][k][nq * 16 + i];
                a1[i] += xv * w1;
                a2[i] += xv * w2;
            }
        }

    int m = (b * CO + c) * L1D + l;
    unsigned short* dst = H0 + (size_t)m * NN + nb + nq * 16;
    v8u16 o0, o1;
#pragma unroll
    for (int i = 0; i < 8; ++i) {
        o0[i] = f2bf(tanhf(a1[i])     * sigmoidf_(a2[i]));
        o1[i] = f2bf(tanhf(a1[8 + i]) * sigmoidf_(a2[8 + i]));
    }
    *reinterpret_cast<v8u16*>(dst)     = o0;
    *reinterpret_cast<v8u16*>(dst + 8) = o1;
}

// --------------------------- diffusion GEMM (WMMA) --------------------------
// D_T[m][w] = sum_v A[w][v] * S_T[m][v]   (all bf16, f32 accum)
__global__ __launch_bounds__(256) void k_diff_gemm(
    const unsigned short* __restrict__ A,    // [2048][2048]
    const unsigned short* __restrict__ Bm,   // [4096][2048]  (rows m, cols v)
    unsigned short* __restrict__ D)          // [4096][2048]  (rows m, cols w)
{
    const int lane = threadIdx.x & 31;
    const int wave = blockIdx.x * 8 + (threadIdx.x >> 5);
    const int mg = wave & 127;          // 128 groups of 32 m
    const int wg = wave >> 7;           // 32 groups of 64 w
    const int row = lane & 15, half = lane >> 4;
    const int wbase = wg * 64, mbase = mg * 32;

    v8f acc[4][2];
#pragma unroll
    for (int i = 0; i < 4; ++i)
#pragma unroll
        for (int j = 0; j < 2; ++j)
#pragma unroll
            for (int r = 0; r < 8; ++r) acc[i][j][r] = 0.0f;

    for (int k0 = 0; k0 < NN; k0 += 32) {
        v16bf af[4];
#pragma unroll
        for (int wt = 0; wt < 4; ++wt) {
            const unsigned short* p = A + (size_t)(wbase + wt * 16 + row) * NN + k0 + half * 16;
            v8bf lo = *reinterpret_cast<const v8bf*>(p);
            v8bf hi = *reinterpret_cast<const v8bf*>(p + 8);
#pragma unroll
            for (int e = 0; e < 8; ++e) { af[wt][e] = lo[e]; af[wt][8 + e] = hi[e]; }
        }
        v16bf bf[2];
#pragma unroll
        for (int mt = 0; mt < 2; ++mt) {
            const unsigned short* p = Bm + (size_t)(mbase + mt * 16 + row) * NN + k0 + half * 16;
            bf[mt] = *reinterpret_cast<const v16bf*>(p);
        }
#pragma unroll
        for (int wt = 0; wt < 4; ++wt)
#pragma unroll
            for (int mt = 0; mt < 2; ++mt)
                acc[wt][mt] = wmma_bf16(af[wt], bf[mt], acc[wt][mt]);
    }

#pragma unroll
    for (int wt = 0; wt < 4; ++wt)
#pragma unroll
        for (int mt = 0; mt < 2; ++mt) {
            int m = mbase + mt * 16 + row;
            int w = wbase + wt * 16 + half * 8;
            v8u16 o;
#pragma unroll
            for (int r = 0; r < 8; ++r) o[r] = f2bf(acc[wt][mt][r]);
            *reinterpret_cast<v8u16*>(D + (size_t)m * NN + w) = o;
        }
}

// --------------------------- pack H_T -> P (transpose) ----------------------
// P[b][l1][n][q*64+c] = H[((b*64+c)*8+l1)][n]
__global__ __launch_bounds__(256) void k_pack(
    const unsigned short* __restrict__ H, unsigned short* __restrict__ P, int q)
{
    __shared__ unsigned short tile[64][64 + 8];
    int bid = blockIdx.x;
    int nb = (bid & 31) * 64; bid >>= 5;
    int l1 = bid & 7;
    int b  = bid >> 3;

    {
        int c = threadIdx.x >> 2, seg = threadIdx.x & 3;
        const unsigned short* p = H + (size_t)((b * CO + c) * L1D + l1) * NN + nb + seg * 16;
#pragma unroll
        for (int t = 0; t < 16; ++t) tile[c][seg * 16 + t] = p[t];
    }
    __syncthreads();
    {
        int n = threadIdx.x >> 2, cs = threadIdx.x & 3;
        unsigned short* out = P + ((size_t)(b * L1D + l1) * NN + nb + n) * C7 + q * 64 + cs * 16;
#pragma unroll
        for (int t = 0; t < 16; ++t) out[t] = tile[cs * 16 + t][n];
    }
}

// --------------------------- pack mlp weights -------------------------------
// Wp[kt][ch][co][kk] = bf16(mlp_w[co][ch*32+kk][kt])
__global__ void k_pack_w(const float* __restrict__ mw, unsigned short* __restrict__ Wp) {
    int idx = blockIdx.x * 256 + threadIdx.x;
    if (idx >= 3 * 14 * 128 * 32) return;
    int kk = idx & 31;
    int t  = idx >> 5;
    int co = t & 127; t >>= 7;
    int ch = t % 14, kt = t / 14;
    Wp[idx] = f2bf(mw[(co * C7 + ch * 32 + kk) * 3 + kt]);
}

// --------------------------- mlp conv GEMM (WMMA) + gate --------------------
// xg[b][l][n][c] = tanh(g[c]) * sigmoid(g[c+64]) ; g = W * im2col(P) + bias
__global__ __launch_bounds__(256) void k_mlp_gemm(
    const unsigned short* __restrict__ P,   // [8][8][2048][448]
    const unsigned short* __restrict__ Wp,  // [3][14][128][32]
    const float* __restrict__ mbias,        // [128]
    float* __restrict__ xg)                 // [8][6][2048][64]
{
    const int lane = threadIdx.x & 31;
    int wave = blockIdx.x * 8 + (threadIdx.x >> 5);
    const int nt2 = wave & 63; wave >>= 6;
    const int ct  = wave & 3;  wave >>= 2;   // co tile pair index (0..3)
    const int b = wave / LL, l = wave % LL;
    const int row = lane & 15, half = lane >> 4;
    const int nbase = nt2 * 32;
    const int co0 = ct * 16;

    v8f accA[2], accB[2];
#pragma unroll
    for (int nt = 0; nt < 2; ++nt)
#pragma unroll
        for (int r = 0; r < 8; ++r) {
            accA[nt][r] = mbias[co0 + half * 8 + r];
            accB[nt][r] = mbias[64 + co0 + half * 8 + r];
        }

    for (int kt = 0; kt < 3; ++kt) {
        const int l1 = l + kt;
        const size_t pb = (size_t)(b * L1D + l1) * NN;
        for (int ch = 0; ch < 14; ++ch) {
            v16bf bf[2];
#pragma unroll
            for (int nt = 0; nt < 2; ++nt) {
                const unsigned short* p =
                    P + (pb + nbase + nt * 16 + row) * C7 + ch * 32 + half * 16;
                bf[nt] = *reinterpret_cast<const v16bf*>(p);
            }
            v16bf aA, aB;
            const unsigned short* wb = Wp + (size_t)(kt * 14 + ch) * 128 * 32;
            {
                const unsigned short* p = wb + (size_t)(co0 + row) * 32 + half * 16;
                v8bf lo = *reinterpret_cast<const v8bf*>(p);
                v8bf hi = *reinterpret_cast<const v8bf*>(p + 8);
#pragma unroll
                for (int e = 0; e < 8; ++e) { aA[e] = lo[e]; aA[8 + e] = hi[e]; }
                p = wb + (size_t)(64 + co0 + row) * 32 + half * 16;
                lo = *reinterpret_cast<const v8bf*>(p);
                hi = *reinterpret_cast<const v8bf*>(p + 8);
#pragma unroll
                for (int e = 0; e < 8; ++e) { aB[e] = lo[e]; aB[8 + e] = hi[e]; }
            }
#pragma unroll
            for (int nt = 0; nt < 2; ++nt) {
                accA[nt] = wmma_bf16(aA, bf[nt], accA[nt]);
                accB[nt] = wmma_bf16(aB, bf[nt], accB[nt]);
            }
        }
    }

#pragma unroll
    for (int nt = 0; nt < 2; ++nt) {
        int n = nbase + nt * 16 + row;
        float* out = xg + ((size_t)(b * LL + l) * NN + n) * CO + co0 + half * 8;
        v8f o;
#pragma unroll
        for (int r = 0; r < 8; ++r)
            o[r] = tanhf(accA[nt][r]) * sigmoidf_(accB[nt][r]);
        *reinterpret_cast<v8f*>(out) = o;
    }
}

// --------------------------- f1 = sum_c xg * c1w ----------------------------
__global__ void k_f1(const float* __restrict__ xg, const float* __restrict__ c1w,
                     float* __restrict__ f1) {
    __shared__ float w[CO];
    if (threadIdx.x < CO) w[threadIdx.x] = c1w[threadIdx.x];
    __syncthreads();
    int idx = blockIdx.x * 256 + threadIdx.x;      // over B*L*N = 98304
    const float* p = xg + (size_t)idx * CO;
    float s = 0.f;
#pragma unroll
    for (int c = 0; c < CO; ++c) s += p[c] * w[c];
    f1[idx] = s;
}

// --------------------------- f2 = sum_n xg * c2w ----------------------------
__global__ void k_f2(const float* __restrict__ xg, const float* __restrict__ c2w,
                     float* __restrict__ f2) {
    int bl = blockIdx.x;                           // b*6+l
    int c = threadIdx.x & 63, part = threadIdx.x >> 6;
    float s = 0.f;
    for (int n = part; n < NN; n += 4)
        s += xg[((size_t)bl * NN + n) * CO + c] * c2w[n];
    __shared__ float red[4][CO];
    red[part][c] = s;
    __syncthreads();
    if (part == 0) {
        float t = red[0][c] + red[1][c] + red[2][c] + red[3][c];
        int b = bl / LL, l = bl % LL;
        f2[(b * CO + c) * LL + l] = t;
    }
}

// --------------------------- u = f1 @ tat_w ---------------------------------
__global__ void k_u(const float* __restrict__ f1, const float* __restrict__ tw,
                    float* __restrict__ u) {
    int bl = blockIdx.x;
    int c = threadIdx.x & 63, part = threadIdx.x >> 6;
    float s = 0.f;
    for (int n = part; n < NN; n += 4)
        s += f1[bl * NN + n] * tw[n * CO + c];
    __shared__ float red[4][CO];
    red[part][c] = s;
    __syncthreads();
    if (part == 0)
        u[bl * CO + c] = red[0][c] + red[1][c] + red[2][c] + red[3][c];
}

// --------------------------- attention (tiny, one block) --------------------
__global__ void k_att(const float* __restrict__ u, const float* __restrict__ f2,
                      const float* __restrict__ tatb, const float* __restrict__ tatv,
                      const float* __restrict__ bng, const float* __restrict__ bnb,
                      float* __restrict__ coefs) {
    __shared__ float s0[BB * LL * LL], s1[BB * LL * LL], mq[LL], rq[LL];
    for (int idx = threadIdx.x; idx < BB * LL * LL; idx += 256) {
        int b = idx / 36, rm = idx % 36, l = rm / LL, q = rm % LL;
        float s = tatb[l * LL + q];
        for (int c = 0; c < CO; ++c)
            s += u[(b * LL + l) * CO + c] * f2[(b * CO + c) * LL + q];
        s0[idx] = sigmoidf_(s);
    }
    __syncthreads();
    for (int idx = threadIdx.x; idx < BB * LL * LL; idx += 256) {
        int b = idx / 36, rm = idx % 36, l = rm / LL, q = rm % LL;
        float s = 0.f;
#pragma unroll
        for (int p = 0; p < LL; ++p) s += tatv[l * LL + p] * s0[b * 36 + p * LL + q];
        s1[idx] = s;
    }
    __syncthreads();
    if (threadIdx.x < LL) {
        int q = threadIdx.x;
        float sm = 0.f, sq = 0.f;
        for (int b = 0; b < BB; ++b)
            for (int l = 0; l < LL; ++l) {
                float v = s1[b * 36 + l * LL + q];
                sm += v; sq += v * v;
            }
        sm /= 48.f;
        float var = sq / 48.f - sm * sm;
        mq[q] = sm; rq[q] = rsqrtf(var + 1e-5f);
    }
    __syncthreads();
    for (int idx = threadIdx.x; idx < BB * LL * LL; idx += 256) {
        int q = idx % LL;
        s0[idx] = (s1[idx] - mq[q]) * rq[q] * bng[q] + bnb[q];
    }
    __syncthreads();
    for (int idx = threadIdx.x; idx < BB * LL; idx += 256) {
        int base = idx * LL;
        float mx = -3.0e38f;
#pragma unroll
        for (int q = 0; q < LL; ++q) mx = fmaxf(mx, s0[base + q]);
        float e[LL], sum = 0.f;
#pragma unroll
        for (int q = 0; q < LL; ++q) { e[q] = __expf(s0[base + q] - mx); sum += e[q]; }
#pragma unroll
        for (int q = 0; q < LL; ++q) coefs[base + q] = e[q] / sum;
    }
}

// --------------------------- residual 1x1 conv into d_out -------------------
__global__ __launch_bounds__(256) void k_residual(
    const float* __restrict__ x, const float* __restrict__ w1,
    const float* __restrict__ b1, float* __restrict__ out)
{
    __shared__ float xs[CIN][64][LL];     // 48KB
    __shared__ float ws[CO][CIN];         // 8KB
    int bid = blockIdx.x;
    int nb = (bid & 31) * 64;
    int b  = bid >> 5;

    for (int e = threadIdx.x; e < CIN * 64 * LL; e += 256) {
        int q = e % LL, rest = e / LL;
        int n = rest & 63, ci = rest >> 6;
        xs[ci][n][q] = x[((size_t)(b * CIN + ci) * NN + nb + n) * TT + (TT - LL) + q];
    }
    for (int e = threadIdx.x; e < CO * CIN; e += 256) ws[e >> 5][e & 31] = w1[e];
    __syncthreads();

    int c = threadIdx.x & 63, ns = threadIdx.x >> 6;
    for (int nn = ns; nn < 64; nn += 4) {
        float acc[LL];
#pragma unroll
        for (int q = 0; q < LL; ++q) acc[q] = b1[c];
        for (int ci = 0; ci < CIN; ++ci) {
            float w = ws[c][ci];
#pragma unroll
            for (int q = 0; q < LL; ++q) acc[q] += xs[ci][nn][q] * w;
        }
        float* o = out + ((size_t)(b * CO + c) * NN + nb + nn) * LL;
#pragma unroll
        for (int q = 0; q < LL; ++q) o[q] = acc[q];
    }
}

// --------------------------- apply attention: out += xg @ t_coef ------------
__global__ __launch_bounds__(256) void k_apply(
    const float* __restrict__ xg, const float* __restrict__ coefs,
    float* __restrict__ out)
{
    __shared__ float cf[36];
    int bid = blockIdx.x;
    int nb = (bid & 63) * 32;
    int b  = bid >> 6;
    if (threadIdx.x < 36) cf[threadIdx.x] = coefs[b * 36 + threadIdx.x];
    __syncthreads();

    int c = threadIdx.x & 63, ns = threadIdx.x >> 6;
    for (int nn = ns; nn < 32; nn += 4) {
        int n = nb + nn;
        float xl[LL];
#pragma unroll
        for (int l = 0; l < LL; ++l)
            xl[l] = xg[((size_t)(b * LL + l) * NN + n) * CO + c];
        float* o = out + ((size_t)(b * CO + c) * NN + n) * LL;
#pragma unroll
        for (int q = 0; q < LL; ++q) {
            float s = 0.f;
#pragma unroll
            for (int l = 0; l < LL; ++l) s += xl[l] * cf[q * LL + l];   // coefs[b][q][l]
            o[q] += s;
        }
    }
}

// --------------------------- per-channel stats (deterministic) --------------
__global__ void k_stats(const float* __restrict__ y, float* __restrict__ stats) {
    int c = blockIdx.x;
    float s = 0.f, sq = 0.f;
    for (int b = 0; b < BB; ++b) {
        const float* p = y + (size_t)(b * CO + c) * NN * LL;
        for (int i = threadIdx.x; i < NN * LL; i += 256) {
            float v = p[i];
            s += v; sq += v * v;
        }
    }
    __shared__ float rs[256], rqv[256];
    rs[threadIdx.x] = s; rqv[threadIdx.x] = sq;
    __syncthreads();
    for (int st = 128; st > 0; st >>= 1) {
        if (threadIdx.x < st) {
            rs[threadIdx.x]  += rs[threadIdx.x + st];
            rqv[threadIdx.x] += rqv[threadIdx.x + st];
        }
        __syncthreads();
    }
    if (threadIdx.x == 0) { stats[c] = rs[0]; stats[64 + c] = rqv[0]; }
}

// --------------------------- final batchnorm --------------------------------
__global__ void k_bnfinal(float* __restrict__ y, const float* __restrict__ stats,
                          const float* __restrict__ g, const float* __restrict__ bb) {
    int idx = blockIdx.x * 256 + threadIdx.x;     // total 6291456
    int c = (idx / (NN * LL)) & 63;
    const float cnt = (float)(BB * NN * LL);
    float m = stats[c] / cnt;
    float var = stats[64 + c] / cnt - m * m;
    y[idx] = (y[idx] - m) * rsqrtf(var + 1e-5f) * g[c] + bb[c];
}

// ============================================================================
extern "C" void kernel_launch(void* const* d_in, const int* in_sizes, int n_in,
                              void* d_out, int out_size, void* d_ws, size_t ws_size,
                              hipStream_t stream) {
    const float* x       = (const float*)d_in[0];
    const float* support = (const float*)d_in[1];
    const float* conv1_w = (const float*)d_in[2];
    const float* conv1_b = (const float*)d_in[3];
    const float* tconv_w = (const float*)d_in[4];
    const float* tconv_b = (const float*)d_in[5];
    const float* mlp_w   = (const float*)d_in[6];
    const float* mlp_b   = (const float*)d_in[7];
    const float* tat_c1w = (const float*)d_in[8];
    const float* tat_c2w = (const float*)d_in[9];
    const float* tat_w   = (const float*)d_in[10];
    const float* tat_b   = (const float*)d_in[11];
    const float* tat_v   = (const float*)d_in[12];
    const float* tat_bng = (const float*)d_in[13];
    const float* tat_bnb = (const float*)d_in[14];
    const float* bn_g    = (const float*)d_in[15];
    const float* bn_b    = (const float*)d_in[16];
    float* out = (float*)d_out;

    char* base = (char*)d_ws;
    size_t off = 0;
    auto alloc = [&](size_t bytes) -> char* {
        char* p = base + off;
        off = (off + bytes + 255) & ~(size_t)255;
        return p;
    };
    unsigned short* supb = (unsigned short*)alloc((size_t)3 * NN * NN * 2);
    unsigned short* H0   = (unsigned short*)alloc((size_t)MD * NN * 2);
    unsigned short* H1   = (unsigned short*)alloc((size_t)MD * NN * 2);
    unsigned short* H2   = (unsigned short*)alloc((size_t)MD * NN * 2);
    unsigned short* P    = (unsigned short*)alloc((size_t)BB * L1D * NN * C7 * 2);
    unsigned short* Wp   = (unsigned short*)alloc((size_t)3 * 14 * 128 * 32 * 2);
    float* xg    = (float*)alloc((size_t)BB * LL * NN * CO * 4);
    float* f1    = (float*)alloc((size_t)BB * LL * NN * 4);
    float* f2    = (float*)alloc((size_t)BB * CO * LL * 4);
    float* u     = (float*)alloc((size_t)BB * LL * CO * 4);
    float* coefs = (float*)alloc((size_t)BB * LL * LL * 4);
    float* stats = (float*)alloc((size_t)128 * 4);
    (void)ws_size; (void)in_sizes; (void)n_in; (void)out_size;

    k_sup2bf<<<6144, 256, 0, stream>>>(support, supb, 3 * NN * NN);
    k_tconv<<<BB * L1D * 32, 256, 0, stream>>>(x, tconv_w, tconv_b, H0);
    k_pack<<<BB * L1D * 32, 256, 0, stream>>>(H0, P, 0);
    k_pack_w<<<672, 256, 0, stream>>>(mlp_w, Wp);

    for (int i = 0; i < 3; ++i) {
        const unsigned short* Ai = supb + (size_t)i * NN * NN;
        k_diff_gemm<<<512, 256, 0, stream>>>(Ai, H0, H1);
        k_pack<<<BB * L1D * 32, 256, 0, stream>>>(H1, P, 2 * i + 1);
        k_diff_gemm<<<512, 256, 0, stream>>>(Ai, H1, H2);
        k_pack<<<BB * L1D * 32, 256, 0, stream>>>(H2, P, 2 * i + 2);
    }

    k_mlp_gemm<<<1536, 256, 0, stream>>>(P, Wp, mlp_b, xg);

    k_f1<<<BB * LL * NN / 256, 256, 0, stream>>>(xg, tat_c1w, f1);
    k_f2<<<BB * LL, 256, 0, stream>>>(xg, tat_c2w, f2);
    k_u<<<BB * LL, 256, 0, stream>>>(f1, tat_w, u);
    k_att<<<1, 256, 0, stream>>>(u, f2, tat_b, tat_v, tat_bng, tat_bnb, coefs);

    k_residual<<<BB * 32, 256, 0, stream>>>(x, conv1_w, conv1_b, out);
    k_apply<<<BB * 64, 256, 0, stream>>>(xg, coefs, out);
    k_stats<<<CO, 256, 0, stream>>>(out, stats);
    k_bnfinal<<<(BB * CO * NN * LL) / 256, 256, 0, stream>>>(out, stats, bn_g, bn_b);
}
